// HeterogeneityAwarePooling_53669911331379
// MI455X (gfx1250) — matrix operand order repaired and checked by reading, not compiled, not run on hardware
//
#include <hip/hip_runtime.h>
#include <hip/hip_bf16.h>

typedef __attribute__((ext_vector_type(16))) _Float16 v16h;
typedef __attribute__((ext_vector_type(8)))  _Float16 v8h;
typedef __attribute__((ext_vector_type(8)))  float    v8f;
typedef __attribute__((ext_vector_type(2)))  float    v2f;
typedef __attribute__((ext_vector_type(2)))  _Float16 h2;

#define N_NODES 500000
#define H       256
#define HETD    3
#define AH      128
#define B_SEG   64
#define KDIM    259      // H + HET
// K padded to 288 = 9 tiles of 32 for f16 WMMA

// ---------------------------------------------------------------------------
// K0: zero the atomic-accumulated scratch (denom[64] + s[64*256])
// ---------------------------------------------------------------------------
__global__ void k_zero(float* __restrict__ p, int n) {
  int i = blockIdx.x * blockDim.x + threadIdx.x;
  if (i < n) p[i] = 0.0f;
}

// ---------------------------------------------------------------------------
// K_pack: repack W1 (fp32 [259,128]) into f16 WMMA-register order so each
// lane's 32-byte B fragment is one contiguous, coalesced global_load_b128 x2.
// Layout: tile (kt,nt) occupies 1KB; within it, slot s = l16*2 + half holds
// 16 halves, element e -> W1[kt*32 + (e&7) + 8*half + 16*(e>>3)][nt*16 + l16].
// ---------------------------------------------------------------------------
__global__ void k_pack(const float* __restrict__ W1, _Float16* __restrict__ pk) {
  int idx = blockIdx.x * blockDim.x + threadIdx.x;   // 9*8*32*16 = 36864
  if (idx >= 9 * 8 * 32 * 16) return;
  int e    = idx & 15;
  int s    = (idx >> 4) & 31;       // slot within tile
  int nt   = (idx >> 9) & 7;
  int kt   = idx >> 12;
  int l16  = s >> 1;
  int half = s & 1;
  int k = kt * 32 + (e & 7) + 8 * half + 16 * (e >> 3);
  int n = nt * 16 + l16;
  float v = (k < KDIM) ? W1[k * AH + n] : 0.0f;
  pk[idx] = (_Float16)v;
}

// ---------------------------------------------------------------------------
// K1: raw = tanh([x;het] @ W1 + b1) @ W2 + b2 ; e = exp(raw) ; denom += e
// 256 threads = 8 waves; block covers 128 rows; each wave owns a 16-row tile.
// A staged in LDS (row-contiguous, stride 40 halves -> conflict-free b128
// reads); B fragments loaded directly from the packed W1 copy (L2-hot).
// ---------------------------------------------------------------------------
__global__ __launch_bounds__(256) void k_gate(
    const float* __restrict__ x, const float* __restrict__ het,
    const long long* __restrict__ batch,
    const _Float16* __restrict__ W1pk, const float* __restrict__ b1,
    const float* __restrict__ W2, const float* __restrict__ b2,
    float* __restrict__ e_out, float* __restrict__ denom)
{
  __shared__ __align__(16) _Float16 Ast[128 * 40]; // 128 rows x 32 k, stride 40
  __shared__ float w2s[AH];
  __shared__ float b1s[AH];
  __shared__ float dloc[B_SEG];

  const int t    = threadIdx.x;
  const int lane = t & 31;
  const int wave = t >> 5;
  const int half = lane >> 4;   // 0 or 1
  const int l16  = lane & 15;
  const int rowBase = blockIdx.x * 128;

  if (t < AH)   { w2s[t] = W2[t]; b1s[t] = b1[t]; }
  if (t < B_SEG) dloc[t] = 0.0f;
  __syncthreads();

  // accumulators seeded with b1; C/D layout: vgpr r -> M=r+8*half, N=l16
  v8f c[8];
#pragma unroll
  for (int nt = 0; nt < 8; ++nt) {
    float bb = b1s[nt * 16 + l16];
    c[nt] = (v8f){bb, bb, bb, bb, bb, bb, bb, bb};
  }

  const int rloc = wave * 16 + l16;
  for (int kt = 0; kt < 9; ++kt) {            // K = 288 (padded)
    __syncthreads();
    // stage A: 128 rows x 32 k as packed half2 -> ds_store_b32, coalesced
#pragma unroll
    for (int i = 0; i < 8; ++i) {
      int p  = i * 256 + t;                   // pair index 0..2047
      int r  = p >> 4;
      int k2 = (p & 15) * 2;
      int g  = rowBase + r; if (g >= N_NODES) g = N_NODES - 1;
      int kg = kt * 32 + k2;
      float v0 = 0.0f, v1 = 0.0f;
      if (kg + 1 < H) {                        // both in x (common path)
        const float2 xv = *(const float2*)(x + (size_t)g * H + kg);
        v0 = xv.x; v1 = xv.y;
      } else {
        if (kg < H) v0 = x[(size_t)g * H + kg];
        else if (kg < KDIM) v0 = het[g * HETD + (kg - H)];
        if (kg + 1 < H) v1 = x[(size_t)g * H + kg + 1];
        else if (kg + 1 < KDIM) v1 = het[g * HETD + (kg + 1 - H)];
      }
      h2 hv; hv.x = (_Float16)v0; hv.y = (_Float16)v1;
      *(h2*)(Ast + r * 40 + k2) = hv;
    }
    __syncthreads();

    // A fragment: two contiguous 16B chunks per lane (ds_load_b128 x2)
    // elements 0-7 -> K = 8*half..8*half+7 ; 8-15 -> K = 16+8*half..
    const v8h a_lo = *(const v8h*)(Ast + rloc * 40 + 8 * half);
    const v8h a_hi = *(const v8h*)(Ast + rloc * 40 + 16 + 8 * half);
    v16h a = __builtin_shufflevector(a_lo, a_hi,
                                     0, 1, 2, 3, 4, 5, 6, 7,
                                     8, 9, 10, 11, 12, 13, 14, 15);
#pragma unroll
    for (int nt = 0; nt < 8; ++nt) {
      // B fragment: 32 contiguous bytes from packed W1 (global_load_b128 x2)
      const v8h* pb = (const v8h*)(W1pk + ((kt * 8 + nt) * 32 + l16 * 2 + half) * 16);
      v16h b = __builtin_shufflevector(pb[0], pb[1],
                                       0, 1, 2, 3, 4, 5, 6, 7,
                                       8, 9, 10, 11, 12, 13, 14, 15);
      c[nt] = __builtin_amdgcn_wmma_f32_16x16x32_f16(
          false, a, false, b, (short)0, c[nt], false, false);
    }
  }

  // tanh (fp32) then dot with W2 (fp32): per-lane partial over its 8 columns
  float part[8];
#pragma unroll
  for (int r = 0; r < 8; ++r) part[r] = 0.0f;
#pragma unroll
  for (int nt = 0; nt < 8; ++nt) {
    float w2v = w2s[nt * 16 + l16];
#pragma unroll
    for (int r = 0; r < 8; ++r)
      part[r] += tanhf(c[nt][r]) * w2v;
  }
  // reduce across the 16 lanes of each half-wave (xor of bits 0..3 stays in half)
#pragma unroll
  for (int r = 0; r < 8; ++r) {
    float v = part[r];
    v += __shfl_xor(v, 1);
    v += __shfl_xor(v, 2);
    v += __shfl_xor(v, 4);
    v += __shfl_xor(v, 8);
    part[r] = v;
  }
  // lanes l16<8 own row m = l16 + 8*half of this wave's tile
  if (l16 < 8) {
    int m = l16 + 8 * half;
    int g = rowBase + wave * 16 + m;
    if (g < N_NODES) {
      float raw = part[l16] + b2[0];
      float ev  = expf(raw);            // |raw| < 28 -> safe without max-shift
      e_out[g]  = ev;
      atomicAdd(&dloc[(int)batch[g]], ev);
    }
  }
  __syncthreads();
  if (t < B_SEG && dloc[t] != 0.0f) atomicAdd(&denom[t], dloc[t]);
}

// ---------------------------------------------------------------------------
// K2: attn = e/denom[batch] (-> output) and s[b] += attn * x  (run-length
// compressed atomics; batch is sorted so ~1-2 segments per 128-node chunk)
// thread t owns column t; block owns 128 consecutive nodes.
// ---------------------------------------------------------------------------
__global__ __launch_bounds__(256) void k_scatter(
    const float* __restrict__ x, const long long* __restrict__ batch,
    const float* __restrict__ e_in, const float* __restrict__ denom,
    float* __restrict__ attn_out, float* __restrict__ s)
{
  __shared__ float attn_s[128];
  __shared__ int   seg_s[128];
  const int t = threadIdx.x;
  const int base = blockIdx.x * 128;

  if (t < 128) {
    int g = base + t;
    if (g < N_NODES) {
      int sg   = (int)batch[g];
      float av = e_in[g] / denom[sg];
      attn_s[t] = av; seg_s[t] = sg;
      attn_out[g] = av;
    } else { attn_s[t] = 0.0f; seg_s[t] = -1; }
  }
  __syncthreads();

  float acc = 0.0f; int cur = -1;
  for (int j = 0; j < 128; ++j) {
    int sg = seg_s[j];
    if (sg != cur) {
      if (cur >= 0) atomicAdd(&s[cur * H + t], acc);
      acc = 0.0f; cur = sg;
    }
    if (sg < 0) break;
    acc += attn_s[j] * x[(size_t)(base + j) * H + t];
  }
  if (cur >= 0) atomicAdd(&s[cur * H + t], acc);
}

// ---------------------------------------------------------------------------
// K3: z = s @ Wv + (segment nonempty ? bv : 0)  -- 64x256x256 fp32 WMMA GEMM
// Single block, 8 waves, each wave computes 8 of the 64 16x16 output tiles.
// ---------------------------------------------------------------------------
__global__ __launch_bounds__(256) void k_zgemm(
    const float* __restrict__ s, const float* __restrict__ Wv,
    const float* __restrict__ bv, const float* __restrict__ denom,
    float* __restrict__ z)
{
  const int t = threadIdx.x;
  const int lane = t & 31, wave = t >> 5;
  const int half = lane >> 4, l16 = lane & 15;

  for (int i = 0; i < 8; ++i) {
    int idx = wave * 8 + i;
    int mt = idx >> 4, nt = idx & 15;
    int n  = nt * 16 + l16;
    v8f c;
#pragma unroll
    for (int r = 0; r < 8; ++r) {
      int m = mt * 16 + r + 8 * half;
      c[r] = (denom[m] > 0.0f) ? bv[n] : 0.0f;   // sum(attn)=1 per nonempty seg
    }
    int mrow = mt * 16 + l16;
    for (int k0 = 0; k0 < H; k0 += 4) {
      // f32 WMMA 16x16x4: A lane L: M=l16, K=e+2*half ; B: K=e+2*half, N=l16
      int ka = k0 + 2 * half;
      v2f a, b;
      a[0] = s[mrow * H + ka];
      a[1] = s[mrow * H + ka + 1];
      b[0] = Wv[ka * H + n];
      b[1] = Wv[(ka + 1) * H + n];
      c = __builtin_amdgcn_wmma_f32_16x16x4_f32(
          false, a, false, b, (short)0, c, false, false);
    }
#pragma unroll
    for (int r = 0; r < 8; ++r) {
      int m = mt * 16 + r + 8 * half;
      z[m * H + n] = c[r];
    }
  }
}

// ---------------------------------------------------------------------------
extern "C" void kernel_launch(void* const* d_in, const int* in_sizes, int n_in,
                              void* d_out, int out_size, void* d_ws, size_t ws_size,
                              hipStream_t stream) {
  const float*     x     = (const float*)d_in[0];
  const float*     het   = (const float*)d_in[1];
  const long long* batch = (const long long*)d_in[2];
  const float*     W1    = (const float*)d_in[3];
  const float*     b1    = (const float*)d_in[4];
  const float*     W2    = (const float*)d_in[5];
  const float*     b2    = (const float*)d_in[6];
  const float*     Wv    = (const float*)d_in[7];
  const float*     bv    = (const float*)d_in[8];

  float* out  = (float*)d_out;
  float* z    = out;                    // [64, 256]
  float* attn = out + B_SEG * H;        // [N]

  float* ws    = (float*)d_ws;
  float* e     = ws;                    // [N]
  float* denom = ws + N_NODES;          // [64]
  float* s     = denom + B_SEG;         // [64, 256]
  _Float16* W1pk = (_Float16*)(s + B_SEG * H);  // 36864 halves, 16B-aligned

  // zero atomic accumulators (denom + s are contiguous)
  k_zero<<<(B_SEG + B_SEG * H + 255) / 256, 256, 0, stream>>>(denom, B_SEG + B_SEG * H);
  // repack W1 into WMMA fragment order (runs once; reused by all 3907 blocks)
  k_pack<<<(9 * 8 * 32 * 16 + 255) / 256, 256, 0, stream>>>(W1, W1pk);

  int nb = (N_NODES + 127) / 128;
  k_gate<<<nb, 256, 0, stream>>>(x, het, batch, W1pk, b1, W2, b2, e, denom);
  k_scatter<<<nb, 256, 0, stream>>>(x, batch, e, denom, attn, s);
  k_zgemm<<<1, 256, 0, stream>>>(s, Wv, bv, denom, z);
}
